// MoDBlock_22333829939447
// MI455X (gfx1250) — matrix operand order repaired and checked by reading, not compiled, not run on hardware
//
#include <hip/hip_runtime.h>
#include <hip/hip_bf16.h>

typedef _Float16 half_t;
typedef __attribute__((ext_vector_type(16))) _Float16 v16h;
typedef __attribute__((ext_vector_type(8)))  _Float16 v8h;
typedef __attribute__((ext_vector_type(8)))  float    v8f;

#define B_  8
#define T_  4096
#define C_  1024
#define H_  16
#define DH_ 64
#define F_  4096
#define K_  512   // capacity = 0.125 * T

// ---------------- WMMA helpers (gfx1250 16x16x32 f16, f32 accum) -------------
// A (16x32): lane L holds row M=L%16; elems e = K[hs*16 + e] (contiguous),
//            hs = L/16.
static __device__ __forceinline__ v16h load_a_frag(const half_t* p, int hs) {
  v8h lo = *(const v8h*)(p + hs * 16);
  v8h hi = *(const v8h*)(p + hs * 16 + 8);
  v16h r;
#pragma unroll
  for (int i = 0; i < 8; ++i) { r[i] = lo[i]; r[8 + i] = hi[i]; }
  return r;
}
// B (32x16): lane L holds column N=L%16; elems e = K[hs*16 + e] (contiguous).
static __device__ __forceinline__ v8f wmma_f16(v16h a, v16h b, v8f c) {
  return __builtin_amdgcn_wmma_f32_16x16x32_f16(false, a, false, b, (short)0, c,
                                                false, false);
}

// Wave computes a 64x64 tile: 4 A-frags x 4 B-frags -> 16 WMMAs per K-step.
// Each fragment is reused 4x (32 FLOP/byte from L2). Bt is column-major
// (Bt[n][k], row stride ldb), already offset to column n0.
// acc[i][j][v] = C(i*16 + v + 8*hs, j*16 + r), r = lane%16, hs = lane/16.
static __device__ __forceinline__ void gemm_tile_64x64(const half_t* A, int lda,
                                                       const half_t* Bt, int ldb,
                                                       int Kd, v8f acc[4][4]) {
  int lane = threadIdx.x & 31;
  int r = lane & 15;
  int hs = lane >> 4;
  const half_t* ap = A + (size_t)r * lda;
  const half_t* bp = Bt + (size_t)r * ldb + hs * 16;
  for (int kd = 0; kd < Kd; kd += 32) {
    v16h af[4], bf[4];
#pragma unroll
    for (int i = 0; i < 4; ++i) {
      af[i] = load_a_frag(ap + (size_t)(i * 16) * lda + kd, hs);
      bf[i] = *(const v16h*)(bp + (size_t)(i * 16) * ldb + kd);
      // speculative prefetch well ahead of the weight stream
      __builtin_prefetch(bp + (size_t)(i * 16) * ldb + kd + 256, 0, 1);
    }
#pragma unroll
    for (int i = 0; i < 4; ++i)
#pragma unroll
      for (int j = 0; j < 4; ++j) acc[i][j] = wmma_f16(af[i], bf[j], acc[i][j]);
  }
}

// ---------------- small utility kernels --------------------------------------
// Tiled f32->f16 transpose: coalesced reads and writes via padded LDS tile.
__global__ __launch_bounds__(256) void k_cvt_t(const float* __restrict__ src,
                                               half_t* __restrict__ dst,
                                               int R, int N) {
  __shared__ half_t tile[32][33];
  int tx = threadIdx.x & 31, ty = threadIdx.x >> 5;  // 8 rows of 32
  int nb = blockIdx.x % (N >> 5);
  int rb = blockIdx.x / (N >> 5);
  int r0 = rb * 32, n0 = nb * 32;
#pragma unroll
  for (int i = 0; i < 32; i += 8)
    tile[ty + i][tx] = (half_t)src[(size_t)(r0 + ty + i) * N + n0 + tx];
  __syncthreads();
#pragma unroll
  for (int i = 0; i < 32; i += 8)
    dst[(size_t)(n0 + ty + i) * R + r0 + tx] = tile[tx][ty + i];
}

__global__ void k_copy(const float4* __restrict__ src, float4* __restrict__ dst,
                       int n) {
  int i = blockIdx.x * 256 + threadIdx.x;
  if (i < n) dst[i] = src[i];
}

__global__ void k_router(const float* __restrict__ x, const float* __restrict__ Wr,
                         float* __restrict__ logits) {
  int gw = blockIdx.x * 8 + (threadIdx.x >> 5);  // token id over B*T
  int lane = threadIdx.x & 31;
  const float* xp = x + (size_t)gw * C_;
  float s = 0.f;
  for (int i = lane; i < C_; i += 32) s += xp[i] * Wr[i];
#pragma unroll
  for (int off = 16; off; off >>= 1) s += __shfl_xor(s, off);
  if (lane == 0) logits[gw] = s;
}

// Per-batch top-K via in-LDS bitonic sort (desc by value, asc index tiebreak),
// then re-sort the K winners ascending by index.
__global__ __launch_bounds__(1024) void k_topk(const float* __restrict__ logits,
                                               int* __restrict__ sel,
                                               float* __restrict__ wsel) {
  __shared__ float sv[T_];
  __shared__ int si[T_];
  int b = blockIdx.x, tid = threadIdx.x;
  const float* lg = logits + (size_t)b * T_;
  for (int i = tid; i < T_; i += 1024) { sv[i] = lg[i]; si[i] = i; }
  __syncthreads();
  for (int k = 2; k <= T_; k <<= 1) {
    for (int j = k >> 1; j > 0; j >>= 1) {
      for (int i = tid; i < T_; i += 1024) {
        int ixj = i ^ j;
        if (ixj > i) {
          bool fwd = ((i & k) == 0);
          float va = sv[i], vb = sv[ixj];
          int ia = si[i], ib = si[ixj];
          bool pba = (vb > va) || (vb == va && ib < ia);  // ixj should precede i
          bool pab = (va > vb) || (va == vb && ia < ib);
          if (fwd ? pba : pab) { sv[i] = vb; sv[ixj] = va; si[i] = ib; si[ixj] = ia; }
        }
      }
      __syncthreads();
    }
  }
  for (int k = 2; k <= K_; k <<= 1) {
    for (int j = k >> 1; j > 0; j >>= 1) {
      for (int i = tid; i < K_; i += 1024) {
        int ixj = i ^ j;
        if (ixj > i) {
          bool fwd = ((i & k) == 0);
          int ia = si[i], ib = si[ixj];
          if (fwd ? (ib < ia) : (ia < ib)) {
            float t = sv[i]; sv[i] = sv[ixj]; sv[ixj] = t;
            int u = si[i]; si[i] = si[ixj]; si[ixj] = u;
          }
        }
      }
      __syncthreads();
    }
  }
  for (int i = tid; i < K_; i += 1024) {
    sel[(size_t)b * K_ + i] = si[i];
    wsel[(size_t)b * K_ + i] = sv[i];
  }
}

static __device__ __forceinline__ float block_reduce_sum256(float v, float* red) {
  int tid = threadIdx.x;
#pragma unroll
  for (int off = 16; off; off >>= 1) v += __shfl_xor(v, off);
  __syncthreads();
  if ((tid & 31) == 0) red[tid >> 5] = v;
  __syncthreads();
  float t = red[0];
#pragma unroll
  for (int w = 1; w < 8; ++w) t += red[w];
  return t;
}

__global__ __launch_bounds__(256) void k_gather_ln1(const float* __restrict__ x,
                                                    const int* __restrict__ sel,
                                                    const float* __restrict__ g,
                                                    float* __restrict__ tok,
                                                    half_t* __restrict__ h16) {
  __shared__ float red[8];
  int t = blockIdx.x;  // [0, B*K)
  int b = t >> 9;
  int row = sel[t];
  const float* xp = x + ((size_t)b * T_ + row) * C_;
  int tid = threadIdx.x;
  float v0[4];
  float s = 0.f;
#pragma unroll
  for (int i = 0; i < 4; ++i) { v0[i] = xp[tid + 256 * i]; s += v0[i]; }
  s = block_reduce_sum256(s, red);
  float mu = s * (1.f / C_);
  float q = 0.f;
#pragma unroll
  for (int i = 0; i < 4; ++i) { float d = v0[i] - mu; q += d * d; }
  q = block_reduce_sum256(q, red);
  float rs = rsqrtf(q * (1.f / C_) + 1e-5f);
  float* tp = tok + (size_t)t * C_;
  half_t* hp = h16 + (size_t)t * C_;
#pragma unroll
  for (int i = 0; i < 4; ++i) {
    int c = tid + 256 * i;
    tp[c] = v0[i];
    hp[c] = (half_t)((v0[i] - mu) * rs * g[c]);
  }
}

__global__ __launch_bounds__(256) void k_ln2(const float* __restrict__ y,
                                             const float* __restrict__ g,
                                             half_t* __restrict__ m16) {
  __shared__ float red[8];
  int t = blockIdx.x;
  const float* yp = y + (size_t)t * C_;
  int tid = threadIdx.x;
  float v0[4];
  float s = 0.f;
#pragma unroll
  for (int i = 0; i < 4; ++i) { v0[i] = yp[tid + 256 * i]; s += v0[i]; }
  s = block_reduce_sum256(s, red);
  float mu = s * (1.f / C_);
  float q = 0.f;
#pragma unroll
  for (int i = 0; i < 4; ++i) { float d = v0[i] - mu; q += d * d; }
  q = block_reduce_sum256(q, red);
  float rs = rsqrtf(q * (1.f / C_) + 1e-5f);
  half_t* mp = m16 + (size_t)t * C_;
#pragma unroll
  for (int i = 0; i < 4; ++i) {
    int c = tid + 256 * i;
    mp[c] = (half_t)((v0[i] - mu) * rs * g[c]);
  }
}

// ---------------- GEMM kernels ----------------------------------------------
// QKV: per batch M=512 (8 tiles of 64), N=3072 (48 tiles of 64), Kd=1024.
__global__ __launch_bounds__(256) void k_gemm_qkv(const half_t* __restrict__ h16,
                                                  const half_t* __restrict__ WT,
                                                  half_t* __restrict__ q16,
                                                  half_t* __restrict__ k16,
                                                  half_t* __restrict__ vT16) {
  // wave-uniform tile id in SGPRs (threadIdx.x>>5 is uniform within a wave)
  int gw = __builtin_amdgcn_readfirstlane(blockIdx.x * 8 + (threadIdx.x >> 5));
  int b = gw / (8 * 48);
  int t = gw % (8 * 48);
  int mt = t / 48, nt = t % 48;
  const half_t* A = h16 + ((size_t)b * K_ + mt * 64) * C_;
  v8f acc[4][4] = {};
  gemm_tile_64x64(A, C_, WT + (size_t)(nt * 64) * C_, C_, C_, acc);
  int lane = threadIdx.x & 31, r = lane & 15, hs = lane >> 4;
#pragma unroll
  for (int j = 0; j < 4; ++j) {
    int nbase = nt * 64 + j * 16;  // 16-aligned, r<16: which/head are uniform
    int which = nbase >> 10;
    int head = (nbase >> 6) & 15;
    int d = j * 16 + r;  // nbase & 63 == j*16 (nt*64 is 64-aligned)
    size_t bh = (size_t)b * H_ + head;
    if (which == 0) {
#pragma unroll
      for (int i = 0; i < 4; ++i)
#pragma unroll
        for (int v = 0; v < 8; ++v) {
          int m = mt * 64 + i * 16 + v + 8 * hs;
          q16[(bh * K_ + m) * DH_ + d] = (half_t)(acc[i][j][v] * 0.125f);
        }
    } else if (which == 1) {
#pragma unroll
      for (int i = 0; i < 4; ++i)
#pragma unroll
        for (int v = 0; v < 8; ++v) {
          int m = mt * 64 + i * 16 + v + 8 * hs;
          k16[(bh * K_ + m) * DH_ + d] = (half_t)acc[i][j][v];
        }
    } else {
#pragma unroll
      for (int i = 0; i < 4; ++i)
#pragma unroll
        for (int v = 0; v < 8; ++v) {
          int m = mt * 64 + i * 16 + v + 8 * hs;
          vT16[(bh * DH_ + d) * K_ + m] = (half_t)acc[i][j][v];  // V transposed
        }
    }
  }
}

// Flash attention: 128 thr = 4 waves; wave w owns 16 q-rows, full DH=64 output.
__global__ __launch_bounds__(128) void k_attn(const half_t* __restrict__ q16,
                                              const half_t* __restrict__ k16,
                                              const half_t* __restrict__ vT16,
                                              half_t* __restrict__ o16) {
  __shared__ half_t Plds[4][16 * 32];
  int wave = __builtin_amdgcn_readfirstlane(threadIdx.x >> 5);
  int lane = threadIdx.x & 31, r = lane & 15, hs = lane >> 4;
  int qt = blockIdx.x & 7;
  int bh = blockIdx.x >> 3;
  int q0 = qt * 64 + wave * 16;
  const half_t* qbase = q16 + ((size_t)bh * K_ + q0 + r) * DH_;
  v16h Qa0 = load_a_frag(qbase, hs);
  v16h Qa1 = load_a_frag(qbase + 32, hs);
  half_t* pl = Plds[wave];
  v8f o[4] = {};
  float mrow[8], lrow[8];
#pragma unroll
  for (int v = 0; v < 8; ++v) { mrow[v] = -1e30f; lrow[v] = 0.f; }
  int kend = qt * 64 + 64;
  for (int kt = 0; kt < kend; kt += 32) {
    // scores: two 16x16 tiles over DH=64 (2 wmma each)
    v8f s0 = {}, s1 = {};
    const half_t* kb = k16 + ((size_t)bh * K_ + kt + r) * DH_;
    {
      v16h b0 = *(const v16h*)(kb + hs * 16);
      v16h b1 = *(const v16h*)(kb + 32 + hs * 16);
      s0 = wmma_f16(Qa0, b0, s0);
      s0 = wmma_f16(Qa1, b1, s0);
      const half_t* kb2 = kb + 16 * DH_;
      v16h c0 = *(const v16h*)(kb2 + hs * 16);
      v16h c1 = *(const v16h*)(kb2 + 32 + hs * 16);
      s1 = wmma_f16(Qa0, c0, s1);
      s1 = wmma_f16(Qa1, c1, s1);
    }
    // online softmax (rows v+8*hs live in 16-lane halves)
#pragma unroll
    for (int v = 0; v < 8; ++v) {
      int qrow = q0 + v + 8 * hs;
      bool m0ok = (kt + r) <= qrow;
      bool m1ok = (kt + 16 + r) <= qrow;
      float a0 = m0ok ? s0[v] : -1e30f;
      float a1 = m1ok ? s1[v] : -1e30f;
      float mx = fmaxf(a0, a1);
#pragma unroll
      for (int off = 1; off <= 8; off <<= 1) mx = fmaxf(mx, __shfl_xor(mx, off));
      float mnew = fmaxf(mrow[v], mx);
      float corr = __expf(mrow[v] - mnew);
      float p0 = m0ok ? __expf(a0 - mnew) : 0.f;
      float p1 = m1ok ? __expf(a1 - mnew) : 0.f;
      float ps = p0 + p1;
#pragma unroll
      for (int off = 1; off <= 8; off <<= 1) ps += __shfl_xor(ps, off);
      lrow[v] = lrow[v] * corr + ps;
      mrow[v] = mnew;
#pragma unroll
      for (int j = 0; j < 4; ++j) o[j][v] *= corr;
      pl[(v + 8 * hs) * 32 + r] = (half_t)p0;
      pl[(v + 8 * hs) * 32 + 16 + r] = (half_t)p1;
    }
    // P (C-layout) -> A-fragment via per-wave LDS (in-order DS pipe)
    v16h Pa = load_a_frag(pl + r * 32, hs);
#pragma unroll
    for (int j = 0; j < 4; ++j) {
      const half_t* vb = vT16 + ((size_t)bh * DH_ + j * 16 + r) * K_ + kt + hs * 16;
      v16h vfrag = *(const v16h*)vb;
      o[j] = wmma_f16(Pa, vfrag, o[j]);
    }
  }
  int b = bh >> 4, h = bh & 15;
#pragma unroll
  for (int j = 0; j < 4; ++j)
#pragma unroll
    for (int v = 0; v < 8; ++v) {
      int m = q0 + v + 8 * hs;
      o16[((size_t)b * K_ + m) * C_ + h * DH_ + j * 16 + r] =
          (half_t)(o[j][v] / lrow[v]);
    }
}

// Wo: per batch M=512 (8x64), N=1024 (16x64), Kd=1024; residual in place.
__global__ __launch_bounds__(256) void k_gemm_wo(const half_t* __restrict__ o16,
                                                 const half_t* __restrict__ WT,
                                                 float* __restrict__ y) {
  int gw = __builtin_amdgcn_readfirstlane(blockIdx.x * 8 + (threadIdx.x >> 5));
  int b = gw / (8 * 16);
  int t = gw % (8 * 16);
  int mt = t / 16, nt = t % 16;
  const half_t* A = o16 + ((size_t)b * K_ + mt * 64) * C_;
  v8f acc[4][4] = {};
  gemm_tile_64x64(A, C_, WT + (size_t)(nt * 64) * C_, C_, C_, acc);
  int lane = threadIdx.x & 31, r = lane & 15, hs = lane >> 4;
#pragma unroll
  for (int i = 0; i < 4; ++i)
#pragma unroll
    for (int j = 0; j < 4; ++j)
#pragma unroll
      for (int v = 0; v < 8; ++v) {
        int m = mt * 64 + i * 16 + v + 8 * hs;
        int n = nt * 64 + j * 16 + r;
        size_t idx = ((size_t)b * K_ + m) * C_ + n;
        y[idx] += acc[i][j][v];  // y = tok + o@Wo (in place)
      }
}

// W1/W3 dual GEMM: 32x64 tile, shared A frags: 2 A + 8 B -> 16 WMMAs/K-step.
__global__ __launch_bounds__(256) void k_gemm_w13(const half_t* __restrict__ m16,
                                                  const half_t* __restrict__ W1T,
                                                  const half_t* __restrict__ W3T,
                                                  half_t* __restrict__ act16) {
  int gw = __builtin_amdgcn_readfirstlane(blockIdx.x * 8 + (threadIdx.x >> 5));
  int b = gw / (16 * 64);
  int t = gw % (16 * 64);
  int mt = t / 64, nt = t % 64;
  int lane = threadIdx.x & 31, r = lane & 15, hs = lane >> 4;
  const half_t* ap = m16 + ((size_t)b * K_ + mt * 32 + r) * C_;
  const half_t* p1 = W1T + (size_t)(nt * 64 + r) * C_ + hs * 16;
  const half_t* p3 = W3T + (size_t)(nt * 64 + r) * C_ + hs * 16;
  v8f u[2][4] = {}, g2[2][4] = {};
  for (int kd = 0; kd < C_; kd += 32) {
    v16h af[2], w1f[4], w3f[4];
#pragma unroll
    for (int i = 0; i < 2; ++i) af[i] = load_a_frag(ap + (size_t)(i * 16) * C_ + kd, hs);
#pragma unroll
    for (int j = 0; j < 4; ++j) {
      size_t ro = (size_t)(j * 16) * C_ + kd;
      w1f[j] = *(const v16h*)(p1 + ro);
      w3f[j] = *(const v16h*)(p3 + ro);
      __builtin_prefetch(p1 + ro + 256, 0, 1);
      __builtin_prefetch(p3 + ro + 256, 0, 1);
    }
#pragma unroll
    for (int i = 0; i < 2; ++i)
#pragma unroll
      for (int j = 0; j < 4; ++j) {
        u[i][j] = wmma_f16(af[i], w1f[j], u[i][j]);
        g2[i][j] = wmma_f16(af[i], w3f[j], g2[i][j]);
      }
  }
#pragma unroll
  for (int i = 0; i < 2; ++i)
#pragma unroll
    for (int j = 0; j < 4; ++j)
#pragma unroll
      for (int v = 0; v < 8; ++v) {
        int m = mt * 32 + i * 16 + v + 8 * hs;
        int n = nt * 64 + j * 16 + r;
        float uu = u[i][j][v], gg = g2[i][j][v];
        float act = uu / (1.f + __expf(-uu)) * gg;  // silu(u) * g
        act16[((size_t)b * K_ + m) * F_ + n] = (half_t)act;
      }
}

// W2: per batch M=512 (8x64), N=1024 (16x64), Kd=4096; fused residual +
// router-weight scale + scatter-add into out (out pre-filled with x).
__global__ __launch_bounds__(256) void k_gemm_w2(const half_t* __restrict__ act16,
                                                 const half_t* __restrict__ WT,
                                                 const float* __restrict__ y,
                                                 const int* __restrict__ sel,
                                                 const float* __restrict__ wsel,
                                                 float* __restrict__ out) {
  int gw = __builtin_amdgcn_readfirstlane(blockIdx.x * 8 + (threadIdx.x >> 5));
  int b = gw / (8 * 16);
  int t = gw % (8 * 16);
  int mt = t / 16, nt = t % 16;
  const half_t* A = act16 + ((size_t)b * K_ + mt * 64) * F_;
  v8f acc[4][4] = {};
  gemm_tile_64x64(A, F_, WT + (size_t)(nt * 64) * F_, F_, F_, acc);
  int lane = threadIdx.x & 31, r = lane & 15, hs = lane >> 4;
#pragma unroll
  for (int i = 0; i < 4; ++i)
#pragma unroll
    for (int j = 0; j < 4; ++j)
#pragma unroll
      for (int v = 0; v < 8; ++v) {
        int m = mt * 64 + i * 16 + v + 8 * hs;
        int n = nt * 64 + j * 16 + r;
        int tk = b * K_ + m;
        float proc = y[(size_t)tk * C_ + n] + acc[i][j][v];
        int srow = sel[tk];
        size_t oi = ((size_t)b * T_ + srow) * C_ + n;
        out[oi] += proc * wsel[tk];  // unique (b,srow,n) per lane-elem
      }
}

// ---------------- host side --------------------------------------------------
extern "C" void kernel_launch(void* const* d_in, const int* in_sizes, int n_in,
                              void* d_out, int out_size, void* d_ws, size_t ws_size,
                              hipStream_t stream) {
  (void)in_sizes; (void)n_in; (void)out_size; (void)ws_size;
  const float* x    = (const float*)d_in[0];
  // d_in[1] = position_ids (unused)
  const float* Wr   = (const float*)d_in[2];
  const float* ln1g = (const float*)d_in[3];
  const float* Wqkv = (const float*)d_in[4];
  const float* Wo   = (const float*)d_in[5];
  const float* ln2g = (const float*)d_in[6];
  const float* W1   = (const float*)d_in[7];
  const float* W3   = (const float*)d_in[8];
  const float* W2   = (const float*)d_in[9];
  float* out = (float*)d_out;

  char* p = (char*)d_ws;
  auto alloc = [&](size_t bytes) -> void* {
    void* r = (void*)p;
    p += (bytes + 255) & ~(size_t)255;
    return r;
  };
  float*  logits = (float*)alloc((size_t)B_ * T_ * 4);
  int*    sel    = (int*)alloc((size_t)B_ * K_ * 4);
  float*  wsel   = (float*)alloc((size_t)B_ * K_ * 4);
  float*  tok    = (float*)alloc((size_t)B_ * K_ * C_ * 4);   // gathered x, then y
  half_t* h16    = (half_t*)alloc((size_t)B_ * K_ * C_ * 2);  // LN1 out, reused for LN2 out
  char*   qreg   = (char*)alloc((size_t)B_ * K_ * F_ * 2);    // 32MB: q|k|vT|o, later act16
  half_t* q16  = (half_t*)qreg;
  half_t* k16  = q16 + (size_t)B_ * H_ * K_ * DH_;
  half_t* vT16 = k16 + (size_t)B_ * H_ * K_ * DH_;
  half_t* o16  = vT16 + (size_t)B_ * H_ * K_ * DH_;
  half_t* act16 = (half_t*)qreg;  // aliases q/k/vT/o (dead by then)
  half_t* WqkvT = (half_t*)alloc((size_t)C_ * 3 * C_ * 2);
  half_t* WoT   = (half_t*)alloc((size_t)C_ * C_ * 2);
  half_t* W1T   = (half_t*)alloc((size_t)C_ * F_ * 2);
  half_t* W3T   = (half_t*)alloc((size_t)C_ * F_ * 2);
  half_t* W2T   = (half_t*)alloc((size_t)F_ * C_ * 2);

  // f32 -> f16, transposed to [N][Kd] so B-fragments are contiguous
  k_cvt_t<<<(C_ / 32) * (3 * C_ / 32), 256, 0, stream>>>(Wqkv, WqkvT, C_, 3 * C_);
  k_cvt_t<<<(C_ / 32) * (C_ / 32), 256, 0, stream>>>(Wo, WoT, C_, C_);
  k_cvt_t<<<(C_ / 32) * (F_ / 32), 256, 0, stream>>>(W1, W1T, C_, F_);
  k_cvt_t<<<(C_ / 32) * (F_ / 32), 256, 0, stream>>>(W3, W3T, C_, F_);
  k_cvt_t<<<(F_ / 32) * (C_ / 32), 256, 0, stream>>>(W2, W2T, F_, C_);

  k_copy<<<(B_ * T_ * C_ / 4 + 255) / 256, 256, 0, stream>>>((const float4*)x,
                                                             (float4*)out,
                                                             B_ * T_ * C_ / 4);
  k_router<<<B_ * T_ / 8, 256, 0, stream>>>(x, Wr, logits);
  k_topk<<<B_, 1024, 0, stream>>>(logits, sel, wsel);
  k_gather_ln1<<<B_ * K_, 256, 0, stream>>>(x, sel, ln1g, tok, h16);
  k_gemm_qkv<<<B_ * 8 * 48 / 8, 256, 0, stream>>>(h16, WqkvT, q16, k16, vT16);
  k_attn<<<B_ * H_ * (K_ / 64), 128, 0, stream>>>(q16, k16, vT16, o16);
  k_gemm_wo<<<B_ * 8 * 16 / 8, 256, 0, stream>>>(o16, WoT, tok);
  k_ln2<<<B_ * K_, 256, 0, stream>>>(tok, ln2g, h16);
  k_gemm_w13<<<B_ * 16 * 64 / 8, 256, 0, stream>>>(h16, W1T, W3T, act16);
  k_gemm_w2<<<B_ * 8 * 16 / 8, 256, 0, stream>>>(act16, W2T, tok, sel, wsel, out);
}
